// Attention_45303315038264
// MI455X (gfx1250) — matrix operand order repaired and checked by reading, not compiled, not run on hardware
//
#include <hip/hip_runtime.h>

// ---------------------------------------------------------------------------
// MHA forward for B=4, L=2048, DM=768, H=12, DH=64 on gfx1250 (wave32, WMMA).
// bf16 WMMA (16x16x32), f32 accumulate, flash-attention with WMMA row-sums,
// TDM (tensor_load_to_lds) double-buffered K/V staging.
// ---------------------------------------------------------------------------

typedef __attribute__((ext_vector_type(16))) __bf16 v16bf;
typedef __attribute__((ext_vector_type(8)))  float  v8f;
typedef unsigned int u32x4 __attribute__((ext_vector_type(4)));
typedef int          i32x8 __attribute__((ext_vector_type(8)));
typedef int          i32x4 __attribute__((ext_vector_type(4)));

#if defined(__AMDGCN__) && __has_builtin(__builtin_amdgcn_tensor_load_to_lds)
#define USE_TDM 1
#else
#define USE_TDM 0
#endif

union Frag16 {
    uint4 q[2];   // 2 x 16B = 16 bf16
    v16bf v;
};

static __device__ __forceinline__ unsigned short cvt_bf16(float f) {
    union { __bf16 h; unsigned short s; } u;
    u.h = (__bf16)f;
    return u.s;
}

static __device__ __forceinline__ unsigned int pack_bf16x2(float lo, float hi) {
    return (unsigned int)cvt_bf16(lo) | ((unsigned int)cvt_bf16(hi) << 16);
}

static __device__ __forceinline__ v8f v8f_zero() {
    v8f z;
    #pragma unroll
    for (int i = 0; i < 8; ++i) z[i] = 0.0f;
    return z;
}

static __device__ __forceinline__ v8f wmma_bf16(const v16bf& a, const v16bf& b, v8f c) {
    return __builtin_amdgcn_wmma_f32_16x16x32_bf16(false, a, false, b, (short)0, c, false, false);
}

// ---------------------------------------------------------------------------
// TDM: DMA a 64x64 bf16 tile (row stride = rowStrideElems in memory) into LDS
// at byte offset ldsByteOff, padding each 128B row with 16B so the LDS row
// stride is 72 elements (144B) -- matching the ds_load_b128 fragment reads.
// D# packing per CDNA5 ISA Tables 8.3/8.4 (group0 128b, group1 256b).
// This toolchain exposes the 6-arg builtin (extra int32x8 group + cpol).
// ---------------------------------------------------------------------------
static __device__ __forceinline__ void tdm_load_tile64x64(unsigned ldsByteOff,
                                                          const unsigned short* g,
                                                          unsigned rowStrideElems) {
#if USE_TDM
    const unsigned long long ga = (unsigned long long)(const void*)g;
    u32x4 g0;
    g0[0] = 1u;                                              // count=1 (valid), user mode
    g0[1] = ldsByteOff;                                      // lds_addr (bytes)
    g0[2] = (unsigned)ga;                                    // global_addr[31:0]
    g0[3] = (unsigned)((ga >> 32) & 0x01FFFFFFu) | (2u << 30); // addr[56:32] | type=2
    i32x8 g1;
    g1[0] = (1 << 16)    // data_size = 1 -> 2 bytes (bf16)
          | (1 << 20)    // pad_enable
          | (4 << 22)    // pad_interval: code 4 = every 32 DWORDs (128B row)
          | (3 << 25);   // pad_amount:   code 3 = 4 DWORDs (16B)  -> 144B stride
    g1[1] = (int)(64u << 16);   // tensor_dim0 = 64  (bits 79:48, low half)
    g1[2] = (int)(64u << 16);   // tensor_dim0 hi=0 | tensor_dim1 = 64 (low half)
    g1[3] = (int)(64u << 16);   // tensor_dim1 hi=0 | tile_dim0 = 64
    g1[4] = 64;                 // tile_dim1 = 64, tile_dim2 = 0
    g1[5] = (int)rowStrideElems;// tensor_dim0_stride (data_size units), low 32
    g1[6] = 0;
    g1[7] = 0;
    i32x4 gz4 = {0, 0, 0, 0};   // groups 2/3: dims 2..4 unused (2D tensor)
    i32x8 gz8 = {0, 0, 0, 0, 0, 0, 0, 0};
    __builtin_amdgcn_tensor_load_to_lds(g0, g1, gz4, gz4, gz8, 0);
#else
    (void)ldsByteOff; (void)g; (void)rowStrideElems;
#endif
}

// ---------------------------------------------------------------------------
// Tiled GEMM: C[M=8192, N=768] = A[8192,768] * W[768,768] + bias
// Block tile 128x64, 8 waves (4x2), each wave 32x32; K-step 64 (8 WMMA/step).
// MODE 0: A = f32 x,    out bf16 [B,H,L,DH]   (Q,K projections)
// MODE 2: A = f32 x,    out bf16 [B,H,DH,L]   (V projection, pre-transposed)
// MODE 1: A = bf16,     out f32  [B*L, DM]    (final projection)
// ---------------------------------------------------------------------------
template <int MODE>
__global__ __launch_bounds__(256) void mha_gemm_kernel(const void* __restrict__ Aptr,
                                                       const float* __restrict__ W,
                                                       const float* __restrict__ bias,
                                                       void* __restrict__ Out) {
    __shared__ unsigned short at[128 * 72];
    __shared__ unsigned short bt[64 * 72];

    const int tid  = threadIdx.x;
    const int lane = tid & 31;
    const int wave = tid >> 5;
    const int wr   = wave & 3;
    const int wc   = wave >> 2;
    const int row0 = blockIdx.x * 128;
    const int col0 = blockIdx.y * 64;
    const int half = lane >> 4;
    const int lr   = lane & 15;

    const float*          Af32 = (const float*)Aptr;
    const unsigned short* Abf  = (const unsigned short*)Aptr;

    v8f acc[2][2];
    #pragma unroll
    for (int i = 0; i < 2; ++i)
        #pragma unroll
        for (int j = 0; j < 2; ++j) acc[i][j] = v8f_zero();

    for (int k0 = 0; k0 < 768; k0 += 64) {
        // ---- stage A tile (128 x 64), vectorized ----
        if (MODE == 1) {
            #pragma unroll
            for (int idx = tid; idx < 128 * 8; idx += 256) {
                const int m = idx >> 3, kq = idx & 7;
                *(uint4*)&at[m * 72 + kq * 8] =
                    *(const uint4*)(Abf + (size_t)(row0 + m) * 768 + k0 + kq * 8);
            }
        } else {
            #pragma unroll
            for (int idx = tid; idx < 128 * 16; idx += 256) {
                const int m = idx >> 4, kq = idx & 15;
                const float4 f = *(const float4*)(Af32 + (size_t)(row0 + m) * 768 + k0 + kq * 4);
                uint2 u;
                u.x = pack_bf16x2(f.x, f.y);
                u.y = pack_bf16x2(f.z, f.w);
                *(uint2*)&at[m * 72 + kq * 4] = u;
            }
        }
        // ---- stage W tile transposed: bt[n][k] ----
        #pragma unroll
        for (int idx = tid; idx < 64 * 16; idx += 256) {
            const int kk = idx >> 4, nq = idx & 15;
            const float4 f = *(const float4*)(W + (size_t)(k0 + kk) * 768 + col0 + nq * 4);
            bt[(nq * 4 + 0) * 72 + kk] = cvt_bf16(f.x);
            bt[(nq * 4 + 1) * 72 + kk] = cvt_bf16(f.y);
            bt[(nq * 4 + 2) * 72 + kk] = cvt_bf16(f.z);
            bt[(nq * 4 + 3) * 72 + kk] = cvt_bf16(f.w);
        }
        // ---- prefetch next K-step tiles while we compute ----
        if (k0 + 64 < 768) {
            __builtin_prefetch(Af32 + (size_t)(row0 + (tid >> 1)) * 768 + k0 + 64, 0, 1);
            __builtin_prefetch(W + (size_t)(k0 + 64 + (tid >> 2)) * 768 + col0, 0, 1);
        }
        __syncthreads();

        #pragma unroll
        for (int kd = 0; kd < 2; ++kd) {
            Frag16 a[2], b[2];
            #pragma unroll
            for (int i = 0; i < 2; ++i) {
                const int m = wr * 32 + i * 16 + lr;
                a[i].q[0] = *(const uint4*)&at[m * 72 + kd * 32 + half * 8];
                a[i].q[1] = *(const uint4*)&at[m * 72 + kd * 32 + 16 + half * 8];
            }
            #pragma unroll
            for (int j = 0; j < 2; ++j) {
                const int n = wc * 32 + j * 16 + lr;
                const uint4* p = (const uint4*)&bt[n * 72 + kd * 32 + half * 8];
                b[j].q[0] = p[0];
                b[j].q[1] = p[2];
            }
            #pragma unroll
            for (int i = 0; i < 2; ++i)
                #pragma unroll
                for (int j = 0; j < 2; ++j)
                    acc[i][j] = wmma_bf16(a[i].v, b[j].v, acc[i][j]);
        }
        __syncthreads();
    }

    // ---- epilogue: bias + store ----
    #pragma unroll
    for (int i = 0; i < 2; ++i)
        #pragma unroll
        for (int j = 0; j < 2; ++j)
            #pragma unroll
            for (int r = 0; r < 8; ++r) {
                const int Rg = row0 + wr * 32 + i * 16 + half * 8 + r;
                const int Cg = col0 + wc * 32 + j * 16 + lr;
                const float val = acc[i][j][r] + bias[Cg];
                const int bI  = Rg >> 11;
                const int seq = Rg & 2047;
                const int h   = Cg >> 6;
                const int d   = Cg & 63;
                if (MODE == 0) {
                    ((unsigned short*)Out)[(((size_t)(bI * 12 + h)) * 2048 + seq) * 64 + d] =
                        cvt_bf16(val);
                } else if (MODE == 2) {
                    ((unsigned short*)Out)[(((size_t)(bI * 12 + h)) * 64 + d) * 2048 + seq] =
                        cvt_bf16(val);
                } else {
                    ((float*)Out)[(size_t)Rg * 768 + Cg] = val;
                }
            }
}

// ---------------------------------------------------------------------------
// Flash attention per (b,h). Block = 128 threads (4 waves), 64 query rows,
// 64-key tiles, TDM double-buffered K/V staging (wave0 -> K, wave1 -> V^T).
// Per tile/wave: 8 WMMA (QK^T) + 2 WMMA (row-sums) + 8 WMMA (PV).
// ---------------------------------------------------------------------------
__global__ __launch_bounds__(128) void mha_flash_attn_kernel(const unsigned short* __restrict__ q,
                                                             const unsigned short* __restrict__ k,
                                                             const unsigned short* __restrict__ vT,
                                                             unsigned short* __restrict__ pf) {
    // [ KT0 | KT1 | VT0 | VT1 | PB(4 waves) ], all rows padded to 72 elements
    constexpr int TILE = 64 * 72;                  // one 64x64 padded tile
    constexpr int KT0 = 0, KT1 = TILE, VT0 = 2 * TILE, VT1 = 3 * TILE, PB = 4 * TILE;
    __shared__ unsigned short smem[4 * TILE + 4 * 16 * 72];

    const int tid  = threadIdx.x;
    const int lane = tid & 31;
    const int wave = tid >> 5;
    const int half = lane >> 4;
    const int lr   = lane & 15;

    const int bh    = blockIdx.y;
    const int bI    = bh / 12;
    const int h     = bh % 12;
    const int qbase = blockIdx.x * 64 + wave * 16;
    const size_t base = (size_t)bh * 2048 * 64;

    const unsigned smemByteBase = (unsigned)(unsigned long long)(const void*)smem;
    const int ktOff[2] = {KT0, KT1};
    const int vtOff[2] = {VT0, VT1};
    unsigned short* pbw = smem + PB + wave * (16 * 72);

    Frag16 ones;
    ones.q[0] = make_uint4(0x3F803F80u, 0x3F803F80u, 0x3F803F80u, 0x3F803F80u);
    ones.q[1] = ones.q[0];

    // ---- load this wave's Q fragments once ----
    Frag16 qf[2];
    {
        const unsigned short* qrow = q + base + (size_t)(qbase + lr) * 64;
        #pragma unroll
        for (int kd = 0; kd < 2; ++kd) {
            qf[kd].q[0] = *(const uint4*)(qrow + kd * 32 + half * 8);
            qf[kd].q[1] = *(const uint4*)(qrow + kd * 32 + 16 + half * 8);
        }
    }

    v8f o[4];
    #pragma unroll
    for (int j = 0; j < 4; ++j) o[j] = v8f_zero();
    v8f rs = v8f_zero();
    float rmax[8];
    #pragma unroll
    for (int r = 0; r < 8; ++r) rmax[r] = -3.4e38f;

    constexpr int NB = 2048 / 64;

#if USE_TDM
    // ---- prologue: DMA tile 0 into buffer 0 ----
    if (wave == 0) tdm_load_tile64x64(smemByteBase + KT0 * 2, k + base, 64);
    if (wave == 1) tdm_load_tile64x64(smemByteBase + VT0 * 2, vT + base, 2048);
    if (wave < 2) __builtin_amdgcn_s_wait_tensorcnt(0);
    __syncthreads();
#endif

    for (int t = 0; t < NB; ++t) {
        const int cur = t & 1;
        const int kb  = t * 64;
        unsigned short* kt = smem + ktOff[cur];
        unsigned short* vt = smem + vtOff[cur];

#if USE_TDM
        // ---- kick DMA for the NEXT tile while we compute this one ----
        if (t + 1 < NB) {
            const int nxt = cur ^ 1;
            if (wave == 0)
                tdm_load_tile64x64(smemByteBase + ktOff[nxt] * 2,
                                   k + base + (size_t)(kb + 64) * 64, 64);
            if (wave == 1)
                tdm_load_tile64x64(smemByteBase + vtOff[nxt] * 2,
                                   vT + base + (size_t)(kb + 64), 2048);
        }
#else
        // ---- fallback: cooperative synchronous staging ----
        #pragma unroll
        for (int idx = tid; idx < 64 * 32; idx += 128) {
            const int kr = idx >> 5, dp = idx & 31;
            ((unsigned int*)&kt[kr * 72])[dp] =
                ((const unsigned int*)(k + base + (size_t)(kb + kr) * 64))[dp];
        }
        #pragma unroll
        for (int idx = tid; idx < 64 * 32; idx += 128) {
            const int d = idx >> 5, kp = idx & 31;
            ((unsigned int*)&vt[d * 72])[kp] =
                ((const unsigned int*)(vT + base + (size_t)d * 2048 + kb))[kp];
        }
        __syncthreads();
#endif

        // ---- S = Q * K^T : four 16x16 n-tiles, K-dim 64 ----
        v8f sv[4];
        #pragma unroll
        for (int nt = 0; nt < 4; ++nt) {
            v8f s = v8f_zero();
            #pragma unroll
            for (int kd = 0; kd < 2; ++kd) {
                Frag16 bk;
                const uint4* p = (const uint4*)&kt[(nt * 16 + lr) * 72 + kd * 32 + half * 8];
                bk.q[0] = p[0];
                bk.q[1] = p[2];
                s = wmma_bf16(qf[kd].v, bk.v, s);
            }
            sv[nt] = s;
        }

        // ---- online softmax: row-max reduce, exp, rescale, spill P ----
        #pragma unroll
        for (int r = 0; r < 8; ++r) {
            float tmx = fmaxf(fmaxf(sv[0][r], sv[1][r]), fmaxf(sv[2][r], sv[3][r]));
            #pragma unroll
            for (int mk = 1; mk < 16; mk <<= 1) tmx = fmaxf(tmx, __shfl_xor(tmx, mk, 32));
            const float nm = fmaxf(rmax[r], tmx);
            const float sc = __expf(rmax[r] - nm);
            rmax[r] = nm;
            rs[r] *= sc;
            #pragma unroll
            for (int j = 0; j < 4; ++j) o[j][r] *= sc;
            const int mrow = half * 8 + r;
            #pragma unroll
            for (int nt = 0; nt < 4; ++nt)
                pbw[mrow * 72 + nt * 16 + lr] = cvt_bf16(__expf(sv[nt][r] - nm));
        }

        // ---- A-fragments of P (K-dim = 64 keys) ----
        Frag16 pa[2];
        #pragma unroll
        for (int kd = 0; kd < 2; ++kd) {
            pa[kd].q[0] = *(const uint4*)&pbw[lr * 72 + kd * 32 + half * 8];
            pa[kd].q[1] = *(const uint4*)&pbw[lr * 72 + kd * 32 + 16 + half * 8];
        }

        // ---- row-sums via WMMA: rs += P * ones ----
        rs = wmma_bf16(pa[0].v, ones.v, rs);
        rs = wmma_bf16(pa[1].v, ones.v, rs);

        // ---- O += P * V ----
        #pragma unroll
        for (int j = 0; j < 4; ++j) {
            #pragma unroll
            for (int kd = 0; kd < 2; ++kd) {
                Frag16 vb;
                const uint4* p = (const uint4*)&vt[(j * 16 + lr) * 72 + kd * 32 + half * 8];
                vb.q[0] = p[0];
                vb.q[1] = p[2];
                o[j] = wmma_bf16(pa[kd].v, vb.v, o[j]);
            }
        }

#if USE_TDM
        if (wave < 2) __builtin_amdgcn_s_wait_tensorcnt(0);
#endif
        __syncthreads();
    }

    // ---- normalize + store prefinal bf16 [B*L, H*DH] ----
    #pragma unroll
    for (int r = 0; r < 8; ++r) {
        const float inv = 1.0f / rs[r];
        const int qr = qbase + half * 8 + r;
        const size_t orow = ((size_t)(bI * 2048 + qr)) * 768 + h * 64;
        #pragma unroll
        for (int j = 0; j < 4; ++j)
            pf[orow + j * 16 + lr] = cvt_bf16(o[j][r] * inv);
    }
}

// ---------------------------------------------------------------------------
// Host-side launch.
// ---------------------------------------------------------------------------
extern "C" void kernel_launch(void* const* d_in, const int* in_sizes, int n_in,
                              void* d_out, int out_size, void* d_ws, size_t ws_size,
                              hipStream_t stream) {
    (void)in_sizes; (void)n_in; (void)out_size; (void)ws_size;

    const float* x  = (const float*)d_in[0];
    const float* Wq = (const float*)d_in[1];
    const float* bq = (const float*)d_in[2];
    const float* bk_ = nullptr; (void)bk_;
    const float* Wk = (const float*)d_in[3];
    const float* bk = (const float*)d_in[4];
    const float* Wv = (const float*)d_in[5];
    const float* bv = (const float*)d_in[6];
    const float* Wd = (const float*)d_in[7];
    const float* bd = (const float*)d_in[8];

    const size_t headElems = (size_t)4 * 12 * 2048 * 64;
    unsigned short* qb  = (unsigned short*)d_ws;
    unsigned short* kb2 = qb + headElems;
    unsigned short* vTb = kb2 + headElems;
    unsigned short* pfb = vTb + headElems;

    const dim3 gemmGrid(64, 12);
    const dim3 gemmBlock(256);

    mha_gemm_kernel<0><<<gemmGrid, gemmBlock, 0, stream>>>(x, Wq, bq, qb);
    mha_gemm_kernel<0><<<gemmGrid, gemmBlock, 0, stream>>>(x, Wk, bk, kb2);
    mha_gemm_kernel<2><<<gemmGrid, gemmBlock, 0, stream>>>(x, Wv, bv, vTb);

    mha_flash_attn_kernel<<<dim3(32, 48), 128, 0, stream>>>(qb, kb2, vTb, pfb);

    mha_gemm_kernel<1><<<gemmGrid, gemmBlock, 0, stream>>>(pfb, Wd, bd, (float*)d_out);
}
